// SelfAttentionMultiHead_48155173323325
// MI455X (gfx1250) — matrix operand order repaired and checked by reading, not compiled, not run on hardware
//
#include <hip/hip_runtime.h>

// ---------------------------------------------------------------------------
// Problem constants (from reference): n=8, c=256, h=w=96 -> s=9216
// ---------------------------------------------------------------------------
#define NB    8
#define CCH   256
#define HW    9216                 // 96*96
#define MTOT  (NB * HW)            // 73728 rows of the (n*s, c) activation
#define NQKV  (3 * CCH)            // 768
#define EPS   1e-5f

typedef __bf16 bf16_t;
typedef __attribute__((ext_vector_type(16))) __bf16 v16bf;
typedef __attribute__((ext_vector_type(8)))  __bf16 v8bf;
typedef __attribute__((ext_vector_type(8)))  float  v8f;

#if __has_builtin(__builtin_amdgcn_sched_barrier)
#define SCHED_FENCE() __builtin_amdgcn_sched_barrier(0)
#else
#define SCHED_FENCE()
#endif

// ----------------------------- bf16 helpers --------------------------------
static __device__ __forceinline__ bf16_t f2bf(float f) {
  unsigned u = __builtin_bit_cast(unsigned, f);
  u += 0x7FFFu + ((u >> 16) & 1u);         // round-to-nearest-even
  unsigned short h = (unsigned short)(u >> 16);
  return __builtin_bit_cast(bf16_t, h);
}
static __device__ __forceinline__ float bf2f(bf16_t b) {
  unsigned short h = __builtin_bit_cast(unsigned short, b);
  unsigned u = ((unsigned)h) << 16;
  return __builtin_bit_cast(float, u);
}

static __device__ __forceinline__ v16bf cat16(v8bf lo, v8bf hi) {
  return __builtin_shufflevector(lo, hi, 0, 1, 2, 3, 4, 5, 6, 7,
                                 8, 9, 10, 11, 12, 13, 14, 15);
}

// --------------------- WMMA fragment builders (wave32) ----------------------
// 16-bit A 16x32 (MxK): lane half h (=lane>>4), m = lane&15.
// elems 0..15 -> K = h*16 + i  (contig)
static __device__ __forceinline__ v16bf load_a_rowmajor(const bf16_t* A, int lda) {
  const int l = threadIdx.x & 31;
  const int m = l & 15, h = l >> 4;
  const bf16_t* row = A + (size_t)m * lda;
  v8bf lo = *(const v8bf*)(row + h * 16);
  v8bf hi = *(const v8bf*)(row + h * 16 + 8);
  return cat16(lo, hi);
}

// Packed-B layout: for tile (kt, nt) of a KxN matrix (K tiles of 32, N tiles
// of 16), lane l = h*16+nn owns elements i=0..15 -> B[kt*32 + h*16 + i][nt*16+nn],
// stored contiguously:  packed[((nt*KT + kt)*32 + l)*16 + i]
static __device__ __forceinline__ v16bf load_b_packed(const bf16_t* packed,
                                                      int kt, int nt, int KT) {
  const int lane = threadIdx.x & 31;
  const bf16_t* p = packed + (((size_t)nt * KT + kt) * 32 + lane) * 16;
  v8bf lo = *(const v8bf*)(p);
  v8bf hi = *(const v8bf*)(p + 8);
  return cat16(lo, hi);
}

static __device__ __forceinline__ v8f wmma_bf16(v16bf a, v16bf b, v8f c) {
  return __builtin_amdgcn_wmma_f32_16x16x32_bf16(
      /*neg_a=*/false, a, /*neg_b=*/false, b,
      /*c_mod=*/(short)0, c, /*reuse_a=*/false, /*reuse_b=*/false);
}

// -------- CDNA5 async global->LDS copy (ASYNCcnt-tracked, inline asm) -------
static __device__ __forceinline__ void async_load_lds_b128(void* lds_dst,
                                                           const void* gsrc) {
  unsigned dst = (unsigned)(size_t)lds_dst;   // low 32 bits = LDS offset
  asm volatile("global_load_async_to_lds_b128 %0, %1, off"
               :: "v"(dst), "v"(gsrc) : "memory");
}
static __device__ __forceinline__ void wait_asynccnt0() {
  asm volatile("s_wait_asynccnt 0x0" ::: "memory");
}

// ---------------------------------------------------------------------------
// Kernel 0: pack an f32 KxN weight into fragment-packed bf16 (one tile/block)
// ---------------------------------------------------------------------------
__global__ void pack_w_kernel(const float* __restrict__ W, int K, int N,
                              bf16_t* __restrict__ out) {
  const int KT = K >> 5;
  const int nt = blockIdx.x / KT;
  const int kt = blockIdx.x % KT;
#pragma unroll
  for (int e = threadIdx.x; e < 512; e += 256) {
    int i = e & 15;
    int lane = e >> 4;                      // 0..31
    int h = lane >> 4, nn = lane & 15;
    int k = kt * 32 + h * 16 + i;
    int n = nt * 16 + nn;
    out[((size_t)(nt * KT + kt) * 32 + lane) * 16 + i] = f2bf(W[(size_t)k * N + n]);
  }
}

// ---------------------------------------------------------------------------
// Kernel 1: per-channel batch stats -> scale/shift with gamma/beta folded in
// ---------------------------------------------------------------------------
__global__ void bn_stats_kernel(const float* __restrict__ inp,
                                const float* __restrict__ gamma,
                                const float* __restrict__ beta,
                                float* __restrict__ scale,
                                float* __restrict__ shift) {
  __shared__ float s1[256];
  __shared__ float s2[256];
  const int ch = blockIdx.x;
  const int t  = threadIdx.x;
  float a = 0.f, b = 0.f;
  for (int bb = 0; bb < NB; ++bb) {
    const float* base = inp + ((size_t)bb * CCH + ch) * HW;
    for (int p = t; p < HW; p += 256) {
      float v = base[p];
      a += v;
      b += v * v;
    }
  }
  s1[t] = a; s2[t] = b;
  __syncthreads();
  for (int st = 128; st > 0; st >>= 1) {
    if (t < st) { s1[t] += s1[t + st]; s2[t] += s2[t + st]; }
    __syncthreads();
  }
  if (t == 0) {
    const float cnt  = (float)(NB * HW);
    float mean = s1[0] / cnt;
    float var  = s2[0] / cnt - mean * mean;
    float sc   = gamma[ch] * rsqrtf(var + EPS);
    scale[ch] = sc;
    shift[ch] = beta[ch] - mean * sc;
  }
}

// ---------------------------------------------------------------------------
// Kernel 2: normalize + transpose (n,c,hw) -> bf16 x_t laid out (n*hw, c)
// ---------------------------------------------------------------------------
__global__ void bn_transpose_kernel(const float* __restrict__ inp,
                                    const float* __restrict__ scale,
                                    const float* __restrict__ shift,
                                    bf16_t* __restrict__ x_t) {
  __shared__ float tile[64][65];
  const int c0 = blockIdx.x * 64;
  const int p0 = blockIdx.y * 64;
  const int bb = blockIdx.z;
  const int t  = threadIdx.x;
  const int tx = t & 63, ty = t >> 6;  // ty in 0..3
#pragma unroll
  for (int cc = ty; cc < 64; cc += 4)
    tile[cc][tx] = inp[((size_t)bb * CCH + c0 + cc) * HW + p0 + tx];
  __syncthreads();
  const int ch = c0 + tx;
  const float sc = scale[ch], sh = shift[ch];
#pragma unroll
  for (int pp = ty; pp < 64; pp += 4) {
    float v = tile[tx][pp] * sc + sh;
    x_t[((size_t)bb * HW + p0 + pp) * CCH + ch] = f2bf(v);
  }
}

// ---------------------------------------------------------------------------
// Kernel 3: QKV GEMM  qkv(MTOT x 768) = x_t(MTOT x 256) @ w_qkv + b
// 8 waves/block; each wave computes 16(M) x 96(N).
// sched_barrier pins all 6 B fragments + next-A in registers before the WMMA
// burst; next iteration's loads hoist into the burst -> 6-deep pipeline.
// ---------------------------------------------------------------------------
__global__ void qkv_gemm_kernel(const bf16_t* __restrict__ x_t,
                                const bf16_t* __restrict__ wq_p,
                                const float* __restrict__ b_qkv,
                                bf16_t* __restrict__ qkv) {
  const int wave = threadIdx.x >> 5;
  const int lane = threadIdx.x & 31;
  const int m0  = blockIdx.x * 16;
  const int ntb = wave * 6;                 // 6 N-tiles (96 cols) per wave
  const int nn  = lane & 15;
  const int mr  = (lane >> 4) * 8;
  const int KT  = CCH / 32;

  v8f c[6];
#pragma unroll
  for (int j = 0; j < 6; ++j) {
    float bias = b_qkv[(ntb + j) * 16 + nn];
#pragma unroll
    for (int r = 0; r < 8; ++r) c[j][r] = bias;
  }

  v16bf a = load_a_rowmajor(x_t + (size_t)m0 * CCH, CCH);
#pragma unroll
  for (int kt = 0; kt < KT; ++kt) {
    v16bf b[6];
#pragma unroll
    for (int j = 0; j < 6; ++j) b[j] = load_b_packed(wq_p, kt, ntb + j, KT);
    int ktn = (kt + 1 < KT) ? kt + 1 : kt;
    v16bf an = load_a_rowmajor(x_t + (size_t)m0 * CCH + ktn * 32, CCH);
    SCHED_FENCE();                          // loads stay above, wmmas below
#pragma unroll
    for (int j = 0; j < 6; ++j) c[j] = wmma_bf16(a, b[j], c[j]);
    a = an;
  }
#pragma unroll
  for (int j = 0; j < 6; ++j)
#pragma unroll
    for (int r = 0; r < 8; ++r)
      qkv[(size_t)(m0 + mr + r) * NQKV + (ntb + j) * 16 + nn] = f2bf(c[j][r]);
}

// ---------------------------------------------------------------------------
// Kernel 4: per-batch kv = k^T @ v (256 x 256, K = 9216), stored in packed-B
// layout for kernel 5.
// Double-buffered ASYNC global->LDS pipeline: while waves run WMMAs on slab
// k (32 s-rows x 512 ch of k||v), the next slab streams into the other LDS
// buffer via global_load_async_to_lds_b128; s_wait_asynccnt + barrier swaps.
// Block: 8 waves over 64(M) x 64(N); wave = 16 x 32 (2 WMMAs / step).
// ---------------------------------------------------------------------------
#define K4_LDSK 520   // 512 + 8 pad, keeps 16B alignment per row
__global__ void kv_gemm_kernel(const bf16_t* __restrict__ qkv,
                               bf16_t* __restrict__ kvp) {
  __shared__ bf16_t lds[2][32 * K4_LDSK];
  const int wave = threadIdx.x >> 5;
  const int lane = threadIdx.x & 31;
  const int wm = wave & 3, wn = wave >> 2;
  const int m0 = blockIdx.x * 64 + wm * 16;   // output row = channel d
  const int n0 = blockIdx.y * 64 + wn * 32;   // output col = channel e
  const int bb = blockIdx.z;
  const int h  = lane >> 4;
  const int nn = lane & 15;

  v8f c[2];
#pragma unroll
  for (int j = 0; j < 2; ++j)
#pragma unroll
    for (int r = 0; r < 8; ++r) c[j][r] = 0.f;

  // stage slab starting at s-row k0 into lds[buf] (32 rows x 512 bf16)
  auto stage = [&](int k0, int buf) {
#pragma unroll
    for (int it = 0; it < 8; ++it) {
      int idx  = it * 256 + (int)threadIdx.x;   // 2048 chunks of 8 bf16
      int row  = idx >> 6;                      // 64 chunks per 512-el row
      int col8 = (idx & 63) * 8;
      const bf16_t* src = qkv + ((size_t)bb * HW + k0 + row) * NQKV + CCH + col8;
      async_load_lds_b128(&lds[buf][row * K4_LDSK + col8], src);
    }
  };

  stage(0, 0);
  wait_asynccnt0();
  __syncthreads();

  int cur = 0;
  for (int k0 = 0; k0 < HW; k0 += 32, cur ^= 1) {
    if (k0 + 32 < HW) stage(k0 + 32, cur ^ 1);   // async fill of other buffer

    const bf16_t* L = &lds[cur][0];
    // A = k^T : A[m=d][kk=srow] = L[srow][d]
    const int d = m0 + nn;
    v8bf alo, ahi;
#pragma unroll
    for (int i = 0; i < 8; ++i) {
      alo[i] = L[(h * 16 + i) * K4_LDSK + d];
      ahi[i] = L[(h * 16 + 8 + i) * K4_LDSK + d];
    }
    v16bf a = cat16(alo, ahi);
    // B = v : B[kk][n=e] = L[kk][256 + e]
#pragma unroll
    for (int j = 0; j < 2; ++j) {
      const int e = 256 + n0 + j * 16 + nn;
      v16bf b;
#pragma unroll
      for (int i = 0; i < 16; ++i)
        b[i] = L[(h * 16 + i) * K4_LDSK + e];
      c[j] = wmma_bf16(a, b, c[j]);
    }

    wait_asynccnt0();       // next slab resident
    __syncthreads();        // all waves done reading cur
  }

  // store into packed-B layout: element (k=m, n) of per-batch 256x256 matrix
#pragma unroll
  for (int j = 0; j < 2; ++j) {
    const int n  = n0 + j * 16 + nn;
    const int nt = n >> 4;
#pragma unroll
    for (int r = 0; r < 8; ++r) {
      int m  = m0 + h * 8 + r;
      int kt = m >> 5;
      int h2 = (m >> 4) & 1;
      int i2 = m & 15;
      kvp[(size_t)bb * (CCH * CCH) +
          ((size_t)(nt * (CCH / 32) + kt) * 32 + h2 * 16 + nn) * 16 + i2] = f2bf(c[j][r]);
    }
  }
}

// ---------------------------------------------------------------------------
// Kernel 5: attn = v + q @ kv (per batch 9216 x 256, K = 256)
// 4 waves/block; each wave 16(M) x 64(N); pipelined like kernel 3.
// ---------------------------------------------------------------------------
__global__ void attn_gemm_kernel(const bf16_t* __restrict__ qkv,
                                 const bf16_t* __restrict__ kvp,
                                 bf16_t* __restrict__ attn) {
  const int wave = threadIdx.x >> 5;
  const int lane = threadIdx.x & 31;
  const int m0  = blockIdx.x * 16;          // global row in [0, MTOT)
  const int ntb = wave * 4;                 // 4 N-tiles (64 cols) per wave
  const int bb  = m0 / HW;                  // tiles never straddle batches
  const int nn  = lane & 15;
  const int mr  = (lane >> 4) * 8;
  const int KT  = CCH / 32;
  const bf16_t* kv_b = kvp + (size_t)bb * (CCH * CCH);

  v8f c[4];
#pragma unroll
  for (int j = 0; j < 4; ++j)               // C = v
#pragma unroll
    for (int r = 0; r < 8; ++r)
      c[j][r] = bf2f(qkv[(size_t)(m0 + mr + r) * NQKV + 2 * CCH + (ntb + j) * 16 + nn]);

  v16bf a = load_a_rowmajor(qkv + (size_t)m0 * NQKV, NQKV);           // q
#pragma unroll
  for (int kt = 0; kt < KT; ++kt) {
    v16bf b[4];
#pragma unroll
    for (int j = 0; j < 4; ++j) b[j] = load_b_packed(kv_b, kt, ntb + j, KT);
    int ktn = (kt + 1 < KT) ? kt + 1 : kt;
    v16bf an = load_a_rowmajor(qkv + (size_t)m0 * NQKV + ktn * 32, NQKV);
    SCHED_FENCE();
#pragma unroll
    for (int j = 0; j < 4; ++j) c[j] = wmma_bf16(a, b[j], c[j]);
    a = an;
  }
#pragma unroll
  for (int j = 0; j < 4; ++j)
#pragma unroll
    for (int r = 0; r < 8; ++r)
      attn[(size_t)(m0 + mr + r) * CCH + (ntb + j) * 16 + nn] = f2bf(c[j][r]);
}

// ---------------------------------------------------------------------------
// Kernel 6: y = attn @ w_proj + b_proj, transpose to (n,c,h,w), + residual
// ---------------------------------------------------------------------------
__global__ void proj_gemm_kernel(const bf16_t* __restrict__ attn,
                                 const bf16_t* __restrict__ wp_p,
                                 const float* __restrict__ b_proj,
                                 const float* __restrict__ inp,
                                 float* __restrict__ out) {
  const int wave = threadIdx.x >> 5;
  const int lane = threadIdx.x & 31;
  const int m0  = blockIdx.x * 16;
  const int ntb = wave * 4;
  const int bb  = m0 / HW;
  const int p0  = m0 - bb * HW;
  const int nn  = lane & 15;
  const int mr  = (lane >> 4) * 8;
  const int KT  = CCH / 32;

  v8f c[4];
#pragma unroll
  for (int j = 0; j < 4; ++j)
#pragma unroll
    for (int r = 0; r < 8; ++r) c[j][r] = 0.f;

  v16bf a = load_a_rowmajor(attn + (size_t)m0 * CCH, CCH);
#pragma unroll
  for (int kt = 0; kt < KT; ++kt) {
    v16bf b[4];
#pragma unroll
    for (int j = 0; j < 4; ++j) b[j] = load_b_packed(wp_p, kt, ntb + j, KT);
    int ktn = (kt + 1 < KT) ? kt + 1 : kt;
    v16bf an = load_a_rowmajor(attn + (size_t)m0 * CCH + ktn * 32, CCH);
    SCHED_FENCE();
#pragma unroll
    for (int j = 0; j < 4; ++j) c[j] = wmma_bf16(a, b[j], c[j]);
    a = an;
  }

#pragma unroll
  for (int j = 0; j < 4; ++j) {
    const int ch   = (ntb + j) * 16 + nn;
    const float bs = b_proj[ch];
#pragma unroll
    for (int r = 0; r < 8; ++r) {
      size_t idx = ((size_t)bb * CCH + ch) * HW + p0 + mr + r;
      out[idx] = c[j][r] + bs + inp[idx];
    }
  }
}

// ---------------------------------------------------------------------------
// Launcher
// ---------------------------------------------------------------------------
extern "C" void kernel_launch(void* const* d_in, const int* in_sizes, int n_in,
                              void* d_out, int out_size, void* d_ws, size_t ws_size,
                              hipStream_t stream) {
  (void)in_sizes; (void)n_in; (void)out_size; (void)ws_size;
  const float* inp      = (const float*)d_in[0];
  const float* bn_gamma = (const float*)d_in[1];
  const float* bn_beta  = (const float*)d_in[2];
  const float* w_qkv    = (const float*)d_in[3];
  const float* b_qkv    = (const float*)d_in[4];
  const float* w_proj   = (const float*)d_in[5];
  const float* b_proj   = (const float*)d_in[6];
  float* out = (float*)d_out;

  // Workspace layout:
  //  [0,4KB)          scale/shift
  //  x_t   bf16 MTOT x 256   (37.75 MB)  -- reused as attn after kernel 3
  //  qkv   bf16 MTOT x 768   (113.25 MB)
  //  kvp   bf16 NB x 256x256 (1 MB, packed-B layout)
  //  wq_p  bf16 256x768 packed (384 KB)
  //  wp_p  bf16 256x256 packed (128 KB)
  float*  scale = (float*)d_ws;
  float*  shift = scale + CCH;
  bf16_t* x_t   = (bf16_t*)((char*)d_ws + 4096);
  bf16_t* qkv   = x_t + (size_t)MTOT * CCH;
  bf16_t* kvp   = qkv + (size_t)MTOT * NQKV;
  bf16_t* wq_p  = kvp + (size_t)NB * CCH * CCH;
  bf16_t* wp_p  = wq_p + (size_t)CCH * NQKV;
  bf16_t* attn  = x_t;  // x_t dead after kernel 3; reuse for attn

  pack_w_kernel<<<(CCH / 32) * (NQKV / 16), 256, 0, stream>>>(w_qkv, CCH, NQKV, wq_p);
  pack_w_kernel<<<(CCH / 32) * (CCH / 16), 256, 0, stream>>>(w_proj, CCH, CCH, wp_p);
  bn_stats_kernel<<<CCH, 256, 0, stream>>>(inp, bn_gamma, bn_beta, scale, shift);
  bn_transpose_kernel<<<dim3(CCH / 64, HW / 64, NB), 256, 0, stream>>>(inp, scale, shift, x_t);
  qkv_gemm_kernel<<<dim3(MTOT / 16), 256, 0, stream>>>(x_t, wq_p, b_qkv, qkv);
  kv_gemm_kernel<<<dim3(CCH / 64, CCH / 64, NB), 256, 0, stream>>>(qkv, kvp);
  attn_gemm_kernel<<<dim3(MTOT / 16), 128, 0, stream>>>(qkv, kvp, attn);
  proj_gemm_kernel<<<dim3(MTOT / 16), 128, 0, stream>>>(attn, wp_p, b_proj, inp, out);
}
